// GRU4Rec_4329327034835
// MI455X (gfx1250) — matrix operand order repaired and checked by reading, not compiled, not run on hardware
//
#include <hip/hip_runtime.h>
#include <hip/hip_bf16.h>
#include <hip/hip_fp16.h>

typedef __attribute__((ext_vector_type(16))) _Float16 v16h;
typedef __attribute__((ext_vector_type(8)))  _Float16 v8h;
typedef __attribute__((ext_vector_type(8)))  float    v8f;

#define B_   512
#define T_   200
#define L_   201
#define H_   128
#define G_   384          // 3*H
#define V_   100001
#define ROWS (B_ * T_)    // 102400

// ---------------------------------------------------------------------------
// gfx1250 async global->LDS copy (16B per lane), tracked by ASYNCcnt.
// VDST VGPR = per-lane LDS byte offset; VADDR = per-lane 64-bit global addr.
// ---------------------------------------------------------------------------
__device__ __forceinline__ void async_copy16(void* lds_ptr, const void* gptr) {
  unsigned int l = (unsigned int)(unsigned long long)lds_ptr;  // low 32 = LDS offset
  asm volatile("global_load_async_to_lds_b128 %0, %1, off"
               :: "v"(l), "v"(gptr) : "memory");
}
__device__ __forceinline__ void wait_async_le6() {
  asm volatile("s_wait_asynccnt 0x6" ::: "memory");
}

// ---------------------------------------------------------------------------
// Kernel 0: convert Wih / Whh (f32, row-major [384][128]) to f16 once.
// ---------------------------------------------------------------------------
__global__ void k_cvt_weights(const float* __restrict__ Wih,
                              const float* __restrict__ Whh,
                              _Float16* __restrict__ wih16,
                              _Float16* __restrict__ whh16) {
  int i = blockIdx.x * blockDim.x + threadIdx.x;
  if (i < G_ * H_) {
    wih16[i] = (_Float16)Wih[i];
    whh16[i] = (_Float16)Whh[i];
  }
}

// ---------------------------------------------------------------------------
// Kernel 1: embedding gather (column of emb_W) + LayerNorm1 -> x16 (f16).
// One wave per (b,t) row; wave32 shfl_xor reduction for mean/var.
// ---------------------------------------------------------------------------
__global__ __launch_bounds__(256) void k_gather_ln1(
    const int* __restrict__ input_seqs,
    const float* __restrict__ embW,
    const float* __restrict__ g1,
    const float* __restrict__ b1,
    _Float16* __restrict__ x16) {
  int lane = threadIdx.x & 31;
  int row  = blockIdx.x * 8 + (threadIdx.x >> 5);     // 0..102399
  int b = row / T_, t = row % T_;
  int tok = input_seqs[b * L_ + t];                   // seqs = input_seqs[:, :-1]

  float v[4], s = 0.f, s2 = 0.f;
#pragma unroll
  for (int j = 0; j < 4; ++j) {
    int h = lane + 32 * j;
    float x = embW[(size_t)h * V_ + tok];             // emb = emb_W.T -> column gather
    v[j] = x; s += x; s2 += x * x;
  }
#pragma unroll
  for (int m = 16; m > 0; m >>= 1) {
    s  += __shfl_xor(s,  m, 32);
    s2 += __shfl_xor(s2, m, 32);
  }
  float mu  = s * (1.f / H_);
  float var = s2 * (1.f / H_) - mu * mu;
  float inv = rsqrtf(var + 1e-8f);
#pragma unroll
  for (int j = 0; j < 4; ++j) {
    int h = lane + 32 * j;
    float y = (v[j] - mu) * inv * g1[h] + b1[h];
    x16[(size_t)row * H_ + h] = (_Float16)y;
  }
}

// ---------------------------------------------------------------------------
// Kernel 2: gi = x @ Wih^T + bih.  (102400 x 128) * (128 x 384) -> f32.
// One wave computes a 16x64 tile (4 col tiles sharing the A registers):
// 16x v_wmma_f32_16x16x32_f16 per wave, A loaded once per k-step.
// ---------------------------------------------------------------------------
__global__ __launch_bounds__(256) void k_gi_gemm(
    const _Float16* __restrict__ x16,
    const _Float16* __restrict__ wih16,
    const float* __restrict__ bih,
    float* __restrict__ gi) {
  int lane = threadIdx.x & 31;
  int wid  = threadIdx.x >> 5;
  int id   = blockIdx.x * 8 + wid;                    // 0 .. 6400*6-1
  int rowTile = id / 6, cgrp = id % 6;
  int rowBase = rowTile * 16, colBase0 = cgrp * 64;
  int n  = lane & 15;
  int hi = lane >> 4;

  v8f c0 = {}, c1 = {}, c2 = {}, c3 = {};
#pragma unroll
  for (int ks = 0; ks < 4; ++ks) {
    // A: 16x32 f16 tile of x. Lane holds row M=n; K=e+16*hi
    const _Float16* arow = x16 + (size_t)(rowBase + n) * H_ + ks * 32;
    v8h a0 = *(const v8h*)(arow + 16 * hi);
    v8h a1 = *(const v8h*)(arow + 16 * hi + 8);
    v16h a;
#pragma unroll
    for (int e = 0; e < 8; ++e) { a[e] = a0[e]; a[8 + e] = a1[e]; }
    // B: 32x16 tiles, B[k][nn] = Wih[col+nn][kbase+k]. Lane holds N=n, K=e+16*hi.
    const _Float16* bbase = wih16 + (size_t)(colBase0 + n) * H_ + ks * 32 + 16 * hi;
    v16h bm0 = *(const v16h*)(bbase);
    v16h bm1 = *(const v16h*)(bbase + 16 * H_);
    v16h bm2 = *(const v16h*)(bbase + 32 * H_);
    v16h bm3 = *(const v16h*)(bbase + 48 * H_);
    c0 = __builtin_amdgcn_wmma_f32_16x16x32_f16(false, a, false, bm0, (short)0, c0, false, false);
    c1 = __builtin_amdgcn_wmma_f32_16x16x32_f16(false, a, false, bm1, (short)0, c1, false, false);
    c2 = __builtin_amdgcn_wmma_f32_16x16x32_f16(false, a, false, bm2, (short)0, c2, false, false);
    c3 = __builtin_amdgcn_wmma_f32_16x16x32_f16(false, a, false, bm3, (short)0, c3, false, false);
  }
  v8f cc[4] = {c0, c1, c2, c3};
#pragma unroll
  for (int j = 0; j < 4; ++j) {
    float bias = bih[colBase0 + j * 16 + n];
#pragma unroll
    for (int r = 0; r < 8; ++r) {
      int m = r + 8 * hi;                              // C/D layout
      gi[(size_t)(rowBase + m) * G_ + colBase0 + j * 16 + n] = cc[j][r] + bias;
    }
  }
}

// ---------------------------------------------------------------------------
// Kernel 3: sequential GRU recurrence + LayerNorm2 + pos/neg dot products.
// 32 blocks x 256 threads; block owns 16 batch rows, loops t = 0..199.
// gi slice for step t+1 is streamed into LDS with global_load_async_to_lds
// (double-buffered) while the WMMA gh GEMM for step t runs.
// ---------------------------------------------------------------------------
__global__ __launch_bounds__(256) void k_gru_ln2_dots(
    const float* __restrict__ gi,
    const _Float16* __restrict__ whh16,
    const float* __restrict__ bhh,
    const float* __restrict__ ln2g,
    const float* __restrict__ ln2b,
    const int* __restrict__ input_seqs,
    const int* __restrict__ negs,
    const float* __restrict__ embW,
    float* __restrict__ out) {
  __shared__ __align__(16) _Float16 hs16[16][136];   // h as f16 (WMMA A), padded
  __shared__ float hs[16][132];                      // h state f32, padded
  __shared__ float gh[16][392];                      // gh staging, padded
  __shared__ __align__(16) float gib[2][16][392];    // gi double buffer, padded

  int tid  = threadIdx.x;
  int lane = tid & 31, wid = tid >> 5;
  int n = lane & 15, hi = lane >> 4;
  int bBase = blockIdx.x * 16;

  for (int i = tid; i < 16 * 136; i += 256) ((_Float16*)hs16)[i] = (_Float16)0.f;
  for (int i = tid; i < 16 * 132; i += 256) ((float*)hs)[i] = 0.f;

  int grow  = tid >> 4;          // gate-phase row 0..15
  int gcol0 = (tid & 15) * 8;    // gate-phase col base (8 contiguous cols)
  int bg = bBase + grow;

  // Precompute this thread's 6 async chunk coordinates (1536 x 16B chunks).
  int c_r[6], c_c[6];
#pragma unroll
  for (int k = 0; k < 6; ++k) {
    int c = tid + 256 * k;
    c_r[k] = (c * 4) / G_;        // row within the 16-row tile
    c_c[k] = (c * 4) % G_;        // col within 384
  }

  // Preload step 0 into buffer 0.
#pragma unroll
  for (int k = 0; k < 6; ++k) {
    const float* gp = gi + ((size_t)(bBase + c_r[k]) * T_ + 0) * G_ + c_c[k];
    async_copy16(&gib[0][c_r[k]][c_c[k]], gp);
  }
  __syncthreads();

  for (int t = 0; t < T_; ++t) {
    // ---- kick off async load of gi for step t+1 into the other buffer ----
    int tn = (t + 1 < T_) ? (t + 1) : (T_ - 1);       // clamp: harmless redundant load
    int nbuf = (t + 1) & 1;
#pragma unroll
    for (int k = 0; k < 6; ++k) {
      const float* gp = gi + ((size_t)(bBase + c_r[k]) * T_ + tn) * G_ + c_c[k];
      async_copy16(&gib[nbuf][c_r[k]][c_c[k]], gp);
    }

    // ---- gh = h @ Whh^T (WMMA), overlapped with the async loads ----
#pragma unroll
    for (int j = 0; j < 3; ++j) {
      int colBase = (wid + 8 * j) * 16;
      v8f acc = {};
#pragma unroll
      for (int ks = 0; ks < 4; ++ks) {
        const _Float16* ap = &hs16[n][ks * 32];
        v8h a0 = *(const v8h*)(ap + 16 * hi);
        v8h a1 = *(const v8h*)(ap + 16 * hi + 8);
        v16h a;
#pragma unroll
        for (int e = 0; e < 8; ++e) { a[e] = a0[e]; a[8 + e] = a1[e]; }
        const _Float16* bp = whh16 + (size_t)(colBase + n) * H_ + ks * 32 + 16 * hi;
        v16h bm = *(const v16h*)bp;
        acc = __builtin_amdgcn_wmma_f32_16x16x32_f16(false, a, false, bm,
                                                     (short)0, acc, false, false);
      }
#pragma unroll
      for (int r = 0; r < 8; ++r) gh[r + 8 * hi][colBase + n] = acc[r];
    }

    // step t's 6 async loads (oldest in flight) retired when ASYNCcnt <= 6
    wait_async_le6();
    __syncthreads();

    // ---- gates + state update (gi read from LDS buffer) ----
    const float* gr = &gib[t & 1][grow][0];
    float hv[8], s = 0.f, s2 = 0.f;
#pragma unroll
    for (int k = 0; k < 8; ++k) {
      int col = gcol0 + k;
      float ir = gr[col];
      float iz = gr[H_ + col];
      float in = gr[2 * H_ + col];
      float hr = gh[grow][col]          + bhh[col];
      float hz = gh[grow][H_ + col]     + bhh[H_ + col];
      float hn = gh[grow][2 * H_ + col] + bhh[2 * H_ + col];
      float r  = 1.f / (1.f + expf(-(ir + hr)));
      float z  = 1.f / (1.f + expf(-(iz + hz)));
      float nn = tanhf(in + r * hn);
      float hold = hs[grow][col];
      float hnew = (1.f - z) * nn + z * hold;
      hs[grow][col]   = hnew;
      hs16[grow][col] = (_Float16)hnew;
      hv[k] = hnew; s += hnew; s2 += hnew * hnew;
    }
    // 16-lane reduction (lanes owning the same row share a half-wave)
#pragma unroll
    for (int m = 8; m > 0; m >>= 1) {
      s  += __shfl_xor(s,  m, 32);
      s2 += __shfl_xor(s2, m, 32);
    }
    float mu  = s * (1.f / H_);
    float var = s2 * (1.f / H_) - mu * mu;
    float inv = rsqrtf(var + 1e-8f);

    // ---- LN2 + pos/neg dots, fused ----
    int ptok = input_seqs[bg * L_ + t + 1];
    int ntok = negs[bg * L_ + t + 1];
    float pd = 0.f, nd = 0.f;
#pragma unroll
    for (int k = 0; k < 8; ++k) {
      int col = gcol0 + k;
      float lnv = (hv[k] - mu) * inv * ln2g[col] + ln2b[col];
      pd += lnv * embW[(size_t)col * V_ + ptok];
      nd += lnv * embW[(size_t)col * V_ + ntok];
    }
#pragma unroll
    for (int m = 8; m > 0; m >>= 1) {
      pd += __shfl_xor(pd, m, 32);
      nd += __shfl_xor(nd, m, 32);
    }
    if ((tid & 15) == 0) {
      out[(size_t)bg * T_ + t]        = pd;   // pos_logits
      out[ROWS + (size_t)bg * T_ + t] = nd;   // neg_logits
    }
    __syncthreads();   // hs16 fully updated before next step's WMMA reads
  }
}

// ---------------------------------------------------------------------------
extern "C" void kernel_launch(void* const* d_in, const int* in_sizes, int n_in,
                              void* d_out, int out_size, void* d_ws, size_t ws_size,
                              hipStream_t stream) {
  const int*   input_seqs = (const int*)d_in[0];
  const int*   negs       = (const int*)d_in[1];
  const float* embW = (const float*)d_in[2];
  const float* Wih  = (const float*)d_in[3];
  const float* Whh  = (const float*)d_in[4];
  const float* bih  = (const float*)d_in[5];
  const float* bhh  = (const float*)d_in[6];
  const float* ln1g = (const float*)d_in[7];
  const float* ln1b = (const float*)d_in[8];
  const float* ln2g = (const float*)d_in[9];
  const float* ln2b = (const float*)d_in[10];
  float* out = (float*)d_out;

  // Workspace layout (all offsets 32B-aligned):
  //   x16   : ROWS*128 f16      =  26,214,400 B
  //   gi    : ROWS*384 f32      = 157,286,400 B
  //   wih16 : 384*128 f16       =      98,304 B
  //   whh16 : 384*128 f16       =      98,304 B
  char* ws = (char*)d_ws;
  _Float16* x16   = (_Float16*)(ws);
  float*    gi    = (float*)   (ws + 26214400);
  _Float16* wih16 = (_Float16*)(ws + 26214400 + 157286400);
  _Float16* whh16 = (_Float16*)(ws + 26214400 + 157286400 + 98304);

  k_cvt_weights<<<192, 256, 0, stream>>>(Wih, Whh, wih16, whh16);
  k_gather_ln1<<<ROWS / 8, 256, 0, stream>>>(input_seqs, embW, ln1g, ln1b, x16);
  k_gi_gemm<<<(ROWS / 16) * 6 / 8, 256, 0, stream>>>(x16, wih16, bih, gi);
  k_gru_ln2_dots<<<32, 256, 0, stream>>>(gi, whh16, bhh, ln2g, ln2b,
                                         input_seqs, negs, embW, out);
}
